// torch_RNN1_6940667150560
// MI455X (gfx1250) — matrix operand order repaired and checked by reading, not compiled, not run on hardware
//
#include <hip/hip_runtime.h>
#include <math.h>

#define BATCH   256
#define TSTEPS  512
#define HIDDEN  256
#define INPUT   2

#define BTILE   16          // batch rows per workgroup (= WMMA M)
#define NWAVES  16          // one 16-wide N tile per wave
#define THREADS (NWAVES * 32)
#define KCHUNKS 8           // 256 / 32
#define HPAD    264         // row pad: 528B/row -> bank-conflict-free ds_load_b128

typedef __attribute__((ext_vector_type(16))) _Float16 v16h;
typedef __attribute__((ext_vector_type(8)))  _Float16 v8h;
typedef __attribute__((ext_vector_type(8)))  float    v8f;

__global__ __launch_bounds__(THREADS)
void rnn_relu_wmma(const float* __restrict__ x,     // [B, T, I]
                   const float* __restrict__ Wih,   // [H, I]
                   const float* __restrict__ Whh,   // [H, H]
                   float* __restrict__ out)         // [B]
{
    __shared__ __align__(16) _Float16 hbuf[2][BTILE][HPAD]; // ping-pong hidden state (f16)
    __shared__ float accum[BTILE];                          // sum of squares for norm

    const int tid    = threadIdx.x;
    const int lane   = tid & 31;
    const int wave   = tid >> 5;
    const int nb     = wave * 16;                 // hidden-column tile owned by this wave
    const int bbase  = blockIdx.x * BTILE;        // batch rows owned by this block
    const int mrow   = lane & 15;
    const int hihalf = (lane >> 4) & 1;           // 0: lanes 0-15, 1: lanes 16-31

    // ---- zero LDS ----
    for (int i = tid; i < 2 * BTILE * HPAD; i += THREADS)
        ((_Float16*)hbuf)[i] = (_Float16)0.0f;
    if (tid < BTILE) accum[tid] = 0.0f;
    __syncthreads();

    // ---- preload W_hh^T (+ W_ih extension) as per-wave B fragments in VGPRs ----
    // B layout (16-bit 32x16 K x N): N = lane%16, element j = K in table order (8*half + j&7 + 16*(j>>3)).
    // B[k][n] = Whh[n][k]  (we need h @ Whh^T).
    v16h w[KCHUNKS + 1];
    #pragma unroll
    for (int c = 0; c < KCHUNKS; ++c) {
        #pragma unroll
        for (int j = 0; j < 16; ++j)
            w[c][j] = (_Float16)Whh[(nb + mrow) * HIDDEN + c * 32 + hihalf * 8 + (j & 7) + (j >> 3) * 16];
    }
    // extension chunk: rows K=0,1 hold the two W_ih columns, rest zero
    #pragma unroll
    for (int j = 0; j < 16; ++j) {
        float v = (!hihalf && j < INPUT) ? Wih[(nb + mrow) * INPUT + j] : 0.0f;
        w[KCHUNKS][j] = (_Float16)v;
    }

    // A-fragment sub-row offsets (16-bit 16x32 layout):
    // lanes 0-15 hold K = base+0..7 and base+16..23 ; lanes 16-31 hold base+8..15, base+24..31
    const int ka0 = hihalf ? 8  : 0;
    const int ka1 = hihalf ? 24 : 16;

    const float* xrow = x + ((size_t)(bbase + mrow) * TSTEPS) * INPUT;
    float2 xc = *(const float2*)xrow;             // x[b, 0, :]

    for (int t = 0; t < TSTEPS; ++t) {
        const int cur = t & 1, nxt = cur ^ 1;
        // prefetch next step's x while we compute (uniform clamp, stays in L2)
        const int tn = (t + 1 < TSTEPS) ? (t + 1) : (TSTEPS - 1);
        float2 xn = *(const float2*)(xrow + (size_t)tn * INPUT);

        // ---- GEMM: [16 x 256] (h, shared in LDS) @ [256 x 16] (w, in VGPRs) ----
        v8f c0 = {}; v8f c1 = {};   // two accumulators -> two independent WMMA chains
        #pragma unroll
        for (int c = 0; c < KCHUNKS; ++c) {
            const _Float16* hrow = &hbuf[cur][mrow][c * 32];
            v8h lo = *(const v8h*)(hrow + ka0);   // ds_load_b128
            v8h hi = *(const v8h*)(hrow + ka1);   // ds_load_b128
            v16h a;
            #pragma unroll
            for (int j = 0; j < 8; ++j) { a[j] = lo[j]; a[8 + j] = hi[j]; }
            if (c & 1)
                c1 = __builtin_amdgcn_wmma_f32_16x16x32_f16(false, a, false, w[c],
                                                            (short)0, c1, false, false);
            else
                c0 = __builtin_amdgcn_wmma_f32_16x16x32_f16(false, a, false, w[c],
                                                            (short)0, c0, false, false);
        }
        // extension chunk: A carries x_t in K=0,1 (lo-half lanes), folds x @ W_ih^T in
        {
            v16h a = {};
            a[0] = hihalf ? (_Float16)0.0f : (_Float16)xc.x;
            a[1] = hihalf ? (_Float16)0.0f : (_Float16)xc.y;
            c0 = __builtin_amdgcn_wmma_f32_16x16x32_f16(false, a, false, w[KCHUNKS],
                                                        (short)0, c0, false, false);
        }

        // ---- combine, ReLU, write h_{t+1}; last step also accumulates squared norm ----
        // C layout: VGPR r, lane: M = r + 8*hihalf, N = nb + mrow
        #pragma unroll
        for (int r = 0; r < 8; ++r) {
            float v = c0[r] + c1[r];
            v = fmaxf(v, 0.0f);
            const int m = r + (hihalf ? 8 : 0);
            hbuf[nxt][m][nb + mrow] = (_Float16)v;
            if (t == TSTEPS - 1)
                atomicAdd(&accum[m], v * v);      // ds_add_f32, once at the end
        }
        xc = xn;
        __syncthreads();                          // one barrier per time step
    }

    if (tid < BTILE)
        out[bbase + tid] = sqrtf(accum[tid]);
}

extern "C" void kernel_launch(void* const* d_in, const int* in_sizes, int n_in,
                              void* d_out, int out_size, void* d_ws, size_t ws_size,
                              hipStream_t stream) {
    const float* x   = (const float*)d_in[0];   // [256, 512, 2]
    const float* Wih = (const float*)d_in[1];   // [256, 2]
    const float* Whh = (const float*)d_in[2];   // [256, 256]
    float* out       = (float*)d_out;           // [256]
    (void)in_sizes; (void)n_in; (void)out_size; (void)d_ws; (void)ws_size;

    rnn_relu_wmma<<<BATCH / BTILE, THREADS, 0, stream>>>(x, Wih, Whh, out);
}
